// NunchakuSDXLFeedForward_3186865734261
// MI455X (gfx1250) — 8-bit matrix operand order repaired and checked by reading, compiled, not run on hardware
//
#include <hip/hip_runtime.h>
#include <math.h>

typedef __attribute__((ext_vector_type(8)))  int      v8i;
typedef __attribute__((ext_vector_type(8)))  float    v8f;
typedef __attribute__((ext_vector_type(16))) _Float16 v16h;

#if defined(__gfx1250__) && __has_builtin(__builtin_amdgcn_global_load_async_to_lds_b128) && \
    __has_builtin(__builtin_amdgcn_global_load_async_to_lds_b32) &&                          \
    __has_builtin(__builtin_amdgcn_s_wait_asynccnt)
#define USE_ASYNC 1
typedef int v4i_vs __attribute__((vector_size(16)));
typedef __attribute__((address_space(1))) v4i_vs glb_v4i;
typedef __attribute__((address_space(3))) v4i_vs lds_v4i;
typedef __attribute__((address_space(1))) int    glb_i32;
typedef __attribute__((address_space(3))) int    lds_i32;

__device__ inline lds_v4i* to_lds128(void* p) {
    return (lds_v4i*)(unsigned int)(unsigned long long)p;
}
__device__ inline glb_v4i* to_glb128(const void* p) {
    return (glb_v4i*)(unsigned long long)p;
}
__device__ inline lds_i32* to_lds32(void* p) {
    return (lds_i32*)(unsigned int)(unsigned long long)p;
}
__device__ inline glb_i32* to_glb32(const void* p) {
    return (glb_i32*)(unsigned long long)p;
}
#else
#define USE_ASYNC 0
#endif

__global__ __launch_bounds__(256)
void quant_rows(const float* __restrict__ src, signed char* __restrict__ q,
                float* __restrict__ scales, int rows, int K) {
    const int gpr  = K >> 6;
    const int g    = blockIdx.x * (blockDim.x >> 5) + (threadIdx.x >> 5);
    const int lane = threadIdx.x & 31;
    if (g >= rows * gpr) return;
    const int row = g / gpr;
    const int grp = g - row * gpr;

    const float* p = src + (size_t)row * K + grp * 64 + lane * 2;
    float2 v = *(const float2*)p;
    float am = fmaxf(fabsf(v.x), fabsf(v.y));
#pragma unroll
    for (int off = 16; off > 0; off >>= 1)
        am = fmaxf(am, __shfl_xor(am, off, 32));
    const float scale = fmaxf(am * (1.0f / 7.0f), 1e-8f);
    const float inv   = 1.0f / scale;
    float qa = fminf(7.0f, fmaxf(-8.0f, rintf(v.x * inv)));
    float qb = fminf(7.0f, fmaxf(-8.0f, rintf(v.y * inv)));
    signed char* qp = q + (size_t)row * K + grp * 64 + lane * 2;
    qp[0] = (signed char)(int)qa;
    qp[1] = (signed char)(int)qb;
    if (lane == 0) scales[(size_t)row * gpr + grp] = scale;
}

__device__ inline v16h pack16(const float* __restrict__ p) {
    float4 f0 = *(const float4*)(p);
    float4 f1 = *(const float4*)(p + 4);
    float4 f2 = *(const float4*)(p + 16);
    float4 f3 = *(const float4*)(p + 20);
    v16h v;
    v[0]  = (_Float16)f0.x; v[1]  = (_Float16)f0.y; v[2]  = (_Float16)f0.z; v[3]  = (_Float16)f0.w;
    v[4]  = (_Float16)f1.x; v[5]  = (_Float16)f1.y; v[6]  = (_Float16)f1.z; v[7]  = (_Float16)f1.w;
    v[8]  = (_Float16)f2.x; v[9]  = (_Float16)f2.y; v[10] = (_Float16)f2.z; v[11] = (_Float16)f2.w;
    v[12] = (_Float16)f3.x; v[13] = (_Float16)f3.y; v[14] = (_Float16)f3.z; v[15] = (_Float16)f3.w;
    return v;
}

__global__ __launch_bounds__(256)
void lora_down_gemm(const float* __restrict__ x, const float* __restrict__ ld,
                    float* __restrict__ t, int K) {
    const int lane  = threadIdx.x & 31;
    const int wave  = threadIdx.x >> 5;
    const int lo    = lane & 15, hi = lane >> 4;
    const int mbase = blockIdx.x * 128 + wave * 16;
    const int mrow  = mbase + lo;
    const int kb    = hi * 8;

    v8f acc0 = {};
    v8f acc1 = {};
    for (int k = 0; k < K; k += 32) {
        v16h a  = pack16(x  + (size_t)mrow * K + k + kb);
        v16h b0 = pack16(ld + (size_t)(lo)      * K + k + kb);
        v16h b1 = pack16(ld + (size_t)(16 + lo) * K + k + kb);
        acc0 = __builtin_amdgcn_wmma_f32_16x16x32_f16(false, a, false, b0, (short)0, acc0, false, false);
        acc1 = __builtin_amdgcn_wmma_f32_16x16x32_f16(false, a, false, b1, (short)0, acc1, false, false);
    }
#pragma unroll
    for (int r = 0; r < 8; ++r) {
        const int row = mbase + 8 * hi + r;
        t[(size_t)row * 32 + lo]      = acc0[r];
        t[(size_t)row * 32 + 16 + lo] = acc1[r];
    }
}

template <bool GEGLU>
__global__ __launch_bounds__(256)
void gemm_w4a4(const signed char* __restrict__ qA, const float* __restrict__ sA,
               const signed char* __restrict__ qB, const float* __restrict__ sB,
               const float* __restrict__ loraX, const float* __restrict__ loraUp,
               float* __restrict__ out, int K, int Nout, int gateOff) {
    __shared__ __align__(16) signed char As[2][128 * 64];
    __shared__ __align__(16) signed char Bs[2][2][64 * 64];
    __shared__ float sS[2][256];

    const int t     = threadIdx.x;
    const int lane  = t & 31;
    const int wave  = t >> 5;
    const int waveM = wave >> 1;
    const int waveN = wave & 1;
    const int lo    = lane & 15, hi = lane >> 4;
    const int m0    = blockIdx.x * 128;
    const int n0    = blockIdx.y * 64;
    const int nG    = K >> 6;

    const int arow0 = t >> 2, ac0 = (t & 3);
    const int arow1 = (t + 256) >> 2, ac1 = (t & 3);
    const int brow  = t >> 2, bc = (t & 3);

    const float* ssrcBase;
    if (t < 128)      ssrcBase = sA + (size_t)(m0 + t) * nG;
    else if (t < 192) ssrcBase = sB + (size_t)(n0 + t - 128) * nG;
    else              ssrcBase = sB + (size_t)((GEGLU ? gateOff : 0) + n0 + t - 192) * nG;

    constexpr int STAGE_CNT = GEGLU ? 5 : 4;

    auto stage = [&](int g, int buf) {
        const signed char* aSrc0 = qA + (size_t)(m0 + arow0) * K + (size_t)g * 64 + ac0 * 16;
        const signed char* aSrc1 = qA + (size_t)(m0 + arow1) * K + (size_t)g * 64 + ac1 * 16;
        const signed char* bSrc0 = qB + (size_t)(n0 + brow) * K + (size_t)g * 64 + bc * 16;
        const signed char* bSrc1 = qB + (size_t)(gateOff + n0 + brow) * K + (size_t)g * 64 + bc * 16;
        const float*       sSrc  = ssrcBase + g;
#if USE_ASYNC
        __builtin_amdgcn_global_load_async_to_lds_b128(to_glb128(aSrc0),
            to_lds128(&As[buf][arow0 * 64 + ac0 * 16]), 0, 0);
        __builtin_amdgcn_global_load_async_to_lds_b128(to_glb128(aSrc1),
            to_lds128(&As[buf][arow1 * 64 + ac1 * 16]), 0, 0);
        __builtin_amdgcn_global_load_async_to_lds_b128(to_glb128(bSrc0),
            to_lds128(&Bs[buf][0][brow * 64 + bc * 16]), 0, 0);
        if (GEGLU)
            __builtin_amdgcn_global_load_async_to_lds_b128(to_glb128(bSrc1),
                to_lds128(&Bs[buf][1][brow * 64 + bc * 16]), 0, 0);
        __builtin_amdgcn_global_load_async_to_lds_b32(to_glb32(sSrc),
            to_lds32(&sS[buf][t]), 0, 0);
#else
        *(uint4*)(As[buf] + arow0 * 64 + ac0 * 16) = *(const uint4*)aSrc0;
        *(uint4*)(As[buf] + arow1 * 64 + ac1 * 16) = *(const uint4*)aSrc1;
        *(uint4*)(Bs[buf][0] + brow * 64 + bc * 16) = *(const uint4*)bSrc0;
        if (GEGLU)
            *(uint4*)(Bs[buf][1] + brow * 64 + bc * 16) = *(const uint4*)bSrc1;
        sS[buf][t] = *sSrc;
        if (g + 1 < nG) {
            __builtin_prefetch(aSrc0 + 64, 0, 1);
            __builtin_prefetch(bSrc0 + 64, 0, 1);
        }
#endif
    };

    v8f accA[2][2] = {};
    v8f accG[2][2] = {};

    stage(0, 0);
    for (int g = 0; g < nG; ++g) {
        const int buf = g & 1;
        if (g + 1 < nG) {
            stage(g + 1, buf ^ 1);
#if USE_ASYNC
            __builtin_amdgcn_s_wait_asynccnt(STAGE_CNT);
#endif
        } else {
#if USE_ASYNC
            __builtin_amdgcn_s_wait_asynccnt(0);
#endif
        }
        __syncthreads();

        const signed char* As_ = As[buf];
        const signed char* Bs0 = Bs[buf][0];
        const signed char* Bs1 = Bs[buf][1];
        const float*       sC  = sS[buf];

        v8i afr[2];
#pragma unroll
        for (int mi = 0; mi < 2; ++mi) {
            const int* ap = (const int*)(As_ + (waveM * 32 + mi * 16 + lo) * 64 + hi * 8);
            v8i a;
            a[0] = ap[0];  a[1] = ap[1];
            a[2] = ap[4];  a[3] = ap[5];
            a[4] = ap[8];  a[5] = ap[9];
            a[6] = ap[12]; a[7] = ap[13];
            afr[mi] = a;
        }
        auto loadB = [&](const signed char* bs, int ni) -> v8i {
            const int* bp = (const int*)(bs + (waveN * 32 + ni * 16 + lo) * 64 + hi * 8);
            v8i b;
            b[0] = bp[0]; b[1] = bp[1]; b[2] = bp[4];  b[3] = bp[5];
            b[4] = bp[8]; b[5] = bp[9]; b[6] = bp[12]; b[7] = bp[13];
            return b;
        };
        v8i bfrA[2], bfrG[2];
        bfrA[0] = loadB(Bs0, 0);
        bfrA[1] = loadB(Bs0, 1);
        if (GEGLU) { bfrG[0] = loadB(Bs1, 0); bfrG[1] = loadB(Bs1, 1); }

        float sav[2][8];
#pragma unroll
        for (int mi = 0; mi < 2; ++mi)
#pragma unroll
            for (int r = 0; r < 8; ++r)
                sav[mi][r] = sC[waveM * 32 + mi * 16 + 8 * hi + r];
        float sbv0[2], sbv1[2];
        sbv0[0] = sC[128 + waveN * 32 + lo];
        sbv0[1] = sC[128 + waveN * 32 + 16 + lo];
        if (GEGLU) { sbv1[0] = sC[192 + waveN * 32 + lo]; sbv1[1] = sC[192 + waveN * 32 + 16 + lo]; }

#pragma unroll
        for (int mi = 0; mi < 2; ++mi)
#pragma unroll
            for (int ni = 0; ni < 2; ++ni) {
                v8i c = {};
                c = __builtin_amdgcn_wmma_i32_16x16x64_iu8(true, afr[mi], true, bfrA[ni], c, false, false);
#pragma unroll
                for (int r = 0; r < 8; ++r)
                    accA[mi][ni][r] = fmaf((float)c[r], sav[mi][r] * sbv0[ni], accA[mi][ni][r]);
                if (GEGLU) {
                    v8i cg = {};
                    cg = __builtin_amdgcn_wmma_i32_16x16x64_iu8(true, afr[mi], true, bfrG[ni], cg, false, false);
#pragma unroll
                    for (int r = 0; r < 8; ++r)
                        accG[mi][ni][r] = fmaf((float)cg[r], sav[mi][r] * sbv1[ni], accG[mi][ni][r]);
                }
            }
        __syncthreads();
    }

#pragma unroll
    for (int mi = 0; mi < 2; ++mi) {
        const int arow = m0 + waveM * 32 + mi * 16 + lo;
        v16h aF = pack16(loraX + (size_t)arow * 32 + hi * 8);
#pragma unroll
        for (int ni = 0; ni < 2; ++ni) {
            const int ncol = n0 + waveN * 32 + ni * 16 + lo;
            v16h bF = pack16(loraUp + (size_t)ncol * 32 + hi * 8);
            accA[mi][ni] = __builtin_amdgcn_wmma_f32_16x16x32_f16(false, aF, false, bF, (short)0,
                                                                  accA[mi][ni], false, false);
            if (GEGLU) {
                v16h bG = pack16(loraUp + (size_t)(gateOff + ncol) * 32 + hi * 8);
                accG[mi][ni] = __builtin_amdgcn_wmma_f32_16x16x32_f16(false, aF, false, bG, (short)0,
                                                                      accG[mi][ni], false, false);
            }
        }
    }

#pragma unroll
    for (int mi = 0; mi < 2; ++mi)
#pragma unroll
        for (int ni = 0; ni < 2; ++ni)
#pragma unroll
            for (int r = 0; r < 8; ++r) {
                const int row = m0 + waveM * 32 + mi * 16 + 8 * hi + r;
                const int col = n0 + waveN * 32 + ni * 16 + lo;
                float v = accA[mi][ni][r];
                if (GEGLU) {
                    const float gt = accG[mi][ni][r];
                    v = v * (0.5f * gt * (1.0f + erff(gt * 0.70710678118654752f)));
                }
                out[(size_t)row * Nout + col] = v;
            }
}

extern "C" void kernel_launch(void* const* d_in, const int* in_sizes, int n_in,
                              void* d_out, int out_size, void* d_ws, size_t ws_size,
                              hipStream_t stream) {
    (void)in_sizes; (void)n_in; (void)out_size; (void)ws_size;
    const float* x   = (const float*)d_in[0];
    const float* W1  = (const float*)d_in[1];
    const float* ld1 = (const float*)d_in[2];
    const float* lu1 = (const float*)d_in[3];
    const float* W2  = (const float*)d_in[4];
    const float* ld2 = (const float*)d_in[5];
    const float* lu2 = (const float*)d_in[6];
    float* out = (float*)d_out;

    const int M = 4 * 4096, D = 1280, INNER = 5120, N1 = 2 * INNER;

    char*  ws  = (char*)d_ws;
    size_t off = 0;
    auto carve = [&](size_t bytes) -> char* {
        char* p = ws + off;
        off += (bytes + 255) & ~(size_t)255;
        return p;
    };
    signed char* qx   = (signed char*)carve((size_t)M * D);
    float*       sx   = (float*)carve((size_t)M * (D / 64) * 4);
    signed char* qw1  = (signed char*)carve((size_t)N1 * D);
    float*       sw1  = (float*)carve((size_t)N1 * (D / 64) * 4);
    signed char* qw2  = (signed char*)carve((size_t)D * INNER);
    float*       sw2  = (float*)carve((size_t)D * (INNER / 64) * 4);
    float*       t1   = (float*)carve((size_t)M * 32 * 4);
    float*       t2   = (float*)carve((size_t)M * 32 * 4);
    float*       gbuf = (float*)carve((size_t)M * INNER * 4);
    signed char* qg   = (signed char*)carve((size_t)M * INNER);
    float*       sg   = (float*)carve((size_t)M * (INNER / 64) * 4);

    { int groups = M * (D / 64);
      quant_rows<<<(groups + 7) / 8, 256, 0, stream>>>(x, qx, sx, M, D); }
    { int groups = N1 * (D / 64);
      quant_rows<<<(groups + 7) / 8, 256, 0, stream>>>(W1, qw1, sw1, N1, D); }
    { int groups = D * (INNER / 64);
      quant_rows<<<(groups + 7) / 8, 256, 0, stream>>>(W2, qw2, sw2, D, INNER); }

    lora_down_gemm<<<M / 128, 256, 0, stream>>>(x, ld1, t1, D);

    {
        dim3 grid(M / 128, INNER / 64);
        gemm_w4a4<true><<<grid, 256, 0, stream>>>(qx, sx, qw1, sw1, t1, lu1,
                                                  gbuf, D, INNER, INNER);
    }

    { int groups = M * (INNER / 64);
      quant_rows<<<(groups + 7) / 8, 256, 0, stream>>>(gbuf, qg, sg, M, INNER); }

    lora_down_gemm<<<M / 128, 256, 0, stream>>>(gbuf, ld2, t2, INNER);

    {
        dim3 grid(M / 128, D / 64);
        gemm_w4a4<false><<<grid, 256, 0, stream>>>(qg, sg, qw2, sw2, t2, lu2,
                                                   out, INNER, D, 0);
    }
}
